// ConditionalGraphKernelNetwork_5428838662519
// MI455X (gfx1250) — matrix operand order repaired and checked by reading, not compiled, not run on hardware
//
#include <hip/hip_runtime.h>
#include <hip/hip_bf16.h>
#include <stdint.h>

// ---------------------------------------------------------------------------
// Problem constants (from the reference)
// ---------------------------------------------------------------------------
#define NN       10000
#define EE       60000
#define BB       4
#define NODE_IN  6
#define EDGE_IN  6
#define COND_IN  10
#define SCALE_IN 4
#define HH       64
#define KWH      128      // KW//2
#define KW       256
#define HSQ      4096     // H*H
#define NUM_LAYERS 3

typedef __attribute__((ext_vector_type(16))) __bf16 v16bf;
typedef __attribute__((ext_vector_type(8)))  float  v8f;

// ---------------------------------------------------------------------------
// Small helpers
// ---------------------------------------------------------------------------
__global__ void zero_f32(float* p, int n) {
    int i = blockIdx.x * 256 + threadIdx.x;
    if (i < n) p[i] = 0.f;
}

__global__ void deg_kernel(const int* edge_index, float* deg, int E) {
    int e = blockIdx.x * 256 + threadIdx.x;
    if (e < E) atomicAdd(&deg[edge_index[E + e]], 1.f);
}

// Transpose + pack f32 [K,N] row-major -> bf16 [N,K] (K-contiguous per column)
__global__ void pack_bt(const float* W, __bf16* Wt, int K, int N) {
    int idx = blockIdx.x * 256 + threadIdx.x;
    if (idx >= K * N) return;
    int k = idx / N, n = idx % N;
    Wt[(size_t)n * K + k] = (__bf16)W[idx];
}

// ---------------------------------------------------------------------------
// Condition encoder: u = relu([cond,scale]@cW1+cb1)@cW2+cb2   (B=4 rows)
// also writes u into the second output slot (d_out + N)
// ---------------------------------------------------------------------------
__global__ void cond_encoder(const float* cond, const float* scale,
                             const float* cW1, const float* cb1,
                             const float* cW2, const float* cb2,
                             float* u, float* out_u) {
    __shared__ float hid[BB][HH];
    int t = threadIdx.x;  // 64
    for (int b = 0; b < BB; ++b) {
        float acc = cb1[t];
        for (int j = 0; j < COND_IN; ++j)  acc += cond[b * COND_IN + j]  * cW1[j * HH + t];
        for (int j = 0; j < SCALE_IN; ++j) acc += scale[b * SCALE_IN + j] * cW1[(COND_IN + j) * HH + t];
        hid[b][t] = fmaxf(acc, 0.f);
    }
    __syncthreads();
    for (int b = 0; b < BB; ++b) {
        float acc = cb2[t];
        for (int i = 0; i < HH; ++i) acc += hid[b][i] * cW2[i * HH + t];
        u[b * HH + t] = acc;
        out_u[b * HH + t] = acc;
    }
}

// ---------------------------------------------------------------------------
// Node encoder: h = relu(x@nW1+nb1)@nW2+nb2   (4 nodes per 256-thread block)
// ---------------------------------------------------------------------------
__global__ void node_encoder(const float* x, const float* nW1, const float* nb1,
                             const float* nW2, const float* nb2, float* h, int N) {
    __shared__ float hid[4][HH];
    __shared__ float sW2[HH * HH];
    int t = threadIdx.x;  // 256
    for (int i = t; i < HH * HH; i += 256) sW2[i] = nW2[i];
    int tx = t & 63, ty = t >> 6;
    int n = blockIdx.x * 4 + ty;
    if (n < N) {
        float acc = nb1[tx];
        for (int j = 0; j < NODE_IN; ++j) acc += x[n * NODE_IN + j] * nW1[j * HH + tx];
        hid[ty][tx] = fmaxf(acc, 0.f);
    }
    __syncthreads();
    if (n < N) {
        float a2 = nb2[tx];
        for (int i = 0; i < HH; ++i) a2 += hid[ty][i] * sW2[i * HH + tx];
        h[(size_t)n * HH + tx] = a2;
    }
}

// ---------------------------------------------------------------------------
// Edge MLP layer 1 (fused concat): k1 = relu([edge_attr, u[batch[src]]]@kW1+kb1)
// 128 threads (one per output col), 32 edges per block. Weights in LDS (~37 KB).
// ---------------------------------------------------------------------------
#define G1_EDGES 32
__global__ void edge_mlp1(const float* edge_attr, const int* edge_index, const int* batch,
                          const float* u, const float* kW1, const float* kb1,
                          __bf16* k1, int E) {
    __shared__ float sW[(EDGE_IN + HH) * KWH];
    __shared__ float sb[KWH];
    __shared__ float su[BB * HH];
    int t = threadIdx.x;  // 128
    for (int i = t; i < (EDGE_IN + HH) * KWH; i += 128) sW[i] = kW1[i];
    sb[t] = kb1[t];
    for (int i = t; i < BB * HH; i += 128) su[i] = u[i];
    __syncthreads();
    int e0 = blockIdx.x * G1_EDGES;
    int e1 = (e0 + G1_EDGES < E) ? (e0 + G1_EDGES) : E;
    for (int e = e0; e < e1; ++e) {
        int s = edge_index[e];        // src = edge_index[0][e]
        int b = batch[s];
        float acc = sb[t];
        for (int j = 0; j < EDGE_IN; ++j) acc += edge_attr[e * EDGE_IN + j] * sW[j * KWH + t];
        const float* ub = &su[b * HH];
        for (int i = 0; i < HH; ++i) acc += ub[i] * sW[(EDGE_IN + i) * KWH + t];
        k1[(size_t)e * KWH + t] = (__bf16)fmaxf(acc, 0.f);
    }
}

// ---------------------------------------------------------------------------
// Generic WMMA bf16 GEMM:  C[E,N] = act(A[E,K] @ Bt[N,K]^T + bias), bf16 out.
// Block = 128 threads = 4 waves. Each wave owns a 32x64 strip of C:
//   2 M-tiles x 4 N-tiles = 8 accumulators; every B fragment is reused by both
//   A fragments (8 WMMAs per 12 b128 loads -> halves L2 B-traffic vs 1 M-tile).
// VGPR layouts follow CDNA5 ISA 7.12.2:
//   A 16x32 bf16: lanes 0-15 (K+0..7 in v0-3, K+16..23 in v4-7),
//                 lanes 16-31 shifted by +8 in K.
//   B 32x16 bf16: column = lane&15; same K layout as A (lanes 16-31 shifted by +8 in K).
// ---------------------------------------------------------------------------
__global__ void wmma_gemm(const __bf16* __restrict__ A, const __bf16* __restrict__ Bt,
                          const float* __restrict__ bias, __bf16* __restrict__ C,
                          int K, int Ncols, int relu) {
    const int lane = threadIdx.x & 31;
    const int wave = threadIdx.x >> 5;
    const int m0 = blockIdx.x * 32;               // two 16-row tiles
    const int n0 = (blockIdx.y * 4 + wave) * 64;
    if (n0 >= Ncols) return;                      // wave-uniform exit: EXEC stays all-1s

    const int kh  = lane >> 4;                    // lane half selects K sub-group
    const int rlo = lane & 15;

    v8f acc[2][4];
#pragma unroll
    for (int mt = 0; mt < 2; ++mt)
#pragma unroll
        for (int t = 0; t < 4; ++t)
            acc[mt][t] = (v8f){0.f,0.f,0.f,0.f,0.f,0.f,0.f,0.f};

    const __bf16* arow0 = A + (size_t)(m0 + rlo) * K + 8 * kh;
    const __bf16* arow1 = arow0 + (size_t)16 * K;
    const __bf16* bcol  = Bt + (size_t)(n0 + rlo) * K + 8 * kh;

    for (int k0 = 0; k0 < K; k0 += 32) {
        union { v16bf v; uint4 u[2]; } a0, a1;
        a0.u[0] = *reinterpret_cast<const uint4*>(arow0 + k0);       // K: k0+8kh .. +7
        a0.u[1] = *reinterpret_cast<const uint4*>(arow0 + k0 + 16);  // K: k0+16+8kh .. +7
        a1.u[0] = *reinterpret_cast<const uint4*>(arow1 + k0);
        a1.u[1] = *reinterpret_cast<const uint4*>(arow1 + k0 + 16);

#pragma unroll
        for (int t = 0; t < 4; ++t) {
            union { v16bf v; uint4 u[2]; } b;
            const __bf16* bp = bcol + (size_t)16 * t * K + k0;
            b.u[0] = *reinterpret_cast<const uint4*>(bp);
            b.u[1] = *reinterpret_cast<const uint4*>(bp + 16);
            acc[0][t] = __builtin_amdgcn_wmma_f32_16x16x32_bf16(
                false, a0.v, false, b.v, (short)0, acc[0][t], false, false);
            acc[1][t] = __builtin_amdgcn_wmma_f32_16x16x32_bf16(
                false, a1.v, false, b.v, (short)0, acc[1][t], false, false);
        }
    }

    // C layout: VGPR r -> M = r + 8*(lane>>4), N = lane&15 (within tile)
    const int mb = kh * 8;
#pragma unroll
    for (int mt = 0; mt < 2; ++mt) {
#pragma unroll
        for (int t = 0; t < 4; ++t) {
            int ncc = n0 + 16 * t + rlo;
            float bv = bias[ncc];
#pragma unroll
            for (int r = 0; r < 8; ++r) {
                float v = acc[mt][t][r] + bv;
                if (relu) v = fmaxf(v, 0.f);
                C[(size_t)(m0 + 16 * mt + mb + r) * Ncols + ncc] = (__bf16)v;
            }
        }
    }
}

// ---------------------------------------------------------------------------
// Message passing: one wave per edge. lane handles output cols 2L, 2L+1.
// w row i is 64 bf16 = 128 B -> each lane does one packed dword load (coalesced).
// h[src] broadcast via wave32 shuffles. Scatter-mean numerator via f32 atomics.
// ---------------------------------------------------------------------------
__global__ void msg_kernel(const float* __restrict__ h, const __bf16* __restrict__ w,
                           const int* __restrict__ edge_index, float* __restrict__ agg, int E) {
    int lane = threadIdx.x & 31;
    int wv   = threadIdx.x >> 5;
    int e = blockIdx.x * 8 + wv;               // 256-thread block = 8 edges
    if (e >= E) return;
    int s = edge_index[e];
    int d = edge_index[E + e];
    float ha = h[(size_t)s * HH + lane];
    float hb = h[(size_t)s * HH + 32 + lane];
    const unsigned int* wr = reinterpret_cast<const unsigned int*>(w + (size_t)e * HSQ);
    __builtin_prefetch(wr + 32 * 64, 0, 1);    // next edge's block -> global_prefetch_b8
    float a0 = 0.f, a1 = 0.f;
#pragma unroll 8
    for (int i = 0; i < HH; ++i) {
        float hs = __shfl((i < 32) ? ha : hb, i & 31, 32);
        unsigned int p = wr[i * 32 + lane];
        a0 = fmaf(hs, __uint_as_float(p << 16), a0);
        a1 = fmaf(hs, __uint_as_float(p & 0xffff0000u), a1);
    }
    atomicAdd(&agg[(size_t)d * HH + 2 * lane],     a0);
    atomicAdd(&agg[(size_t)d * HH + 2 * lane + 1], a1);
}

// ---------------------------------------------------------------------------
// Node update: h' = relu(agg/deg + h@root + bias)
// ---------------------------------------------------------------------------
__global__ void update_kernel(const float* agg, const float* deg, const float* hin,
                              const float* root, const float* bias, float* hout, int N) {
    __shared__ float sroot[HH * HH];
    int t = threadIdx.x;  // 256
    for (int i = t; i < HH * HH; i += 256) sroot[i] = root[i];
    __syncthreads();
    int tx = t & 63, ty = t >> 6;
    int n = blockIdx.x * 4 + ty;
    if (n >= N) return;
    float acc = bias[tx] + agg[(size_t)n * HH + tx] / fmaxf(deg[n], 1.f);
    for (int i = 0; i < HH; ++i) acc += hin[(size_t)n * HH + i] * sroot[i * HH + tx];
    hout[(size_t)n * HH + tx] = fmaxf(acc, 0.f);
}

__global__ void head_kernel(const float* h, const float* oW, const float* ob, float* out, int N) {
    int n = blockIdx.x * 256 + threadIdx.x;
    if (n >= N) return;
    float acc = ob[0];
    for (int i = 0; i < HH; ++i) acc += h[(size_t)n * HH + i] * oW[i];
    out[n] = acc;
}

// ---------------------------------------------------------------------------
// Launch
// ---------------------------------------------------------------------------
static inline size_t align256(size_t x) { return (x + 255) & ~size_t(255); }

extern "C" void kernel_launch(void* const* d_in, const int* in_sizes, int n_in,
                              void* d_out, int out_size, void* d_ws, size_t ws_size,
                              hipStream_t stream) {
    const float* x         = (const float*)d_in[0];
    const float* edge_attr = (const float*)d_in[1];
    const float* cond      = (const float*)d_in[2];
    const float* scl       = (const float*)d_in[3];
    const int*   edge_index= (const int*)  d_in[4];
    const int*   batch     = (const int*)  d_in[5];
    const float* nW1 = (const float*)d_in[6],  *nb1 = (const float*)d_in[7];
    const float* nW2 = (const float*)d_in[8],  *nb2 = (const float*)d_in[9];
    const float* cW1 = (const float*)d_in[10], *cb1 = (const float*)d_in[11];
    const float* cW2 = (const float*)d_in[12], *cb2 = (const float*)d_in[13];
    const float* kW1 = (const float*)d_in[14], *kb1 = (const float*)d_in[15];
    const float* kW2 = (const float*)d_in[16], *kb2 = (const float*)d_in[17];
    const float* kW3 = (const float*)d_in[18], *kb3 = (const float*)d_in[19];
    const float* root= (const float*)d_in[20], *cbias=(const float*)d_in[21];
    const float* oW  = (const float*)d_in[22], *ob  = (const float*)d_in[23];
    float* out = (float*)d_out;                  // [N] then u [B*H]

    // ---- workspace carve-up (all 256B aligned) ----
    uint8_t* ws = (uint8_t*)d_ws;
    size_t off = 0;
    float* u_ws = (float*)(ws + off); off = align256(off + BB * HH * sizeof(float));
    float* hA   = (float*)(ws + off); off = align256(off + (size_t)NN * HH * sizeof(float));
    float* hB   = (float*)(ws + off); off = align256(off + (size_t)NN * HH * sizeof(float));
    float* agg  = (float*)(ws + off); off = align256(off + (size_t)NN * HH * sizeof(float));
    float* deg  = (float*)(ws + off); off = align256(off + (size_t)NN * sizeof(float));
    __bf16* k1  = (__bf16*)(ws + off); off = align256(off + (size_t)EE * KWH * sizeof(__bf16));
    __bf16* k2  = (__bf16*)(ws + off); off = align256(off + (size_t)EE * KW  * sizeof(__bf16));
    __bf16* k2t = (__bf16*)(ws + off); off = align256(off + (size_t)KW  * KWH * sizeof(__bf16));
    __bf16* k3t = (__bf16*)(ws + off); off = align256(off + (size_t)HSQ * KW  * sizeof(__bf16));
    __bf16* w   = (__bf16*)(ws + off); off = align256(off + (size_t)EE * HSQ * sizeof(__bf16));
    (void)ws_size; (void)in_sizes; (void)n_in; (void)out_size;

    // ---- init ----
    zero_f32<<<(NN + 255) / 256, 256, 0, stream>>>(deg, NN);
    deg_kernel<<<(EE + 255) / 256, 256, 0, stream>>>(edge_index, deg, EE);

    // ---- encoders ----
    cond_encoder<<<1, HH, 0, stream>>>(cond, scl, cW1, cb1, cW2, cb2, u_ws, out + NN);
    node_encoder<<<(NN + 3) / 4, 256, 0, stream>>>(x, nW1, nb1, nW2, nb2, hA, NN);

    // ---- pack kernel-net weights for WMMA B-tiles (N-major, K-contiguous bf16) ----
    pack_bt<<<(KWH * KW + 255) / 256, 256, 0, stream>>>(kW2, k2t, KWH, KW);
    pack_bt<<<(KW * HSQ + 255) / 256, 256, 0, stream>>>(kW3, k3t, KW, HSQ);

    // ---- kernel network ----
    edge_mlp1<<<(EE + G1_EDGES - 1) / G1_EDGES, KWH, 0, stream>>>(
        edge_attr, edge_index, batch, u_ws, kW1, kb1, k1, EE);

    // layer 2: [E,128] @ [128,256] + relu
    {
        dim3 grid(EE / 32, (KW + 255) / 256);
        wmma_gemm<<<grid, 128, 0, stream>>>(k1, k2t, kb2, k2, KWH, KW, 1);
    }
    // layer 3: [E,256] @ [256,4096] -> per-edge weights w (bf16)
    {
        dim3 grid(EE / 32, (HSQ + 255) / 256);
        wmma_gemm<<<grid, 128, 0, stream>>>(k2, k3t, kb3, w, KW, HSQ, 0);
    }

    // ---- 3 rounds of NNConv message passing ----
    float* hcur = hA;
    float* hnxt = hB;
    for (int layer = 0; layer < NUM_LAYERS; ++layer) {
        zero_f32<<<(NN * HH + 255) / 256, 256, 0, stream>>>(agg, NN * HH);
        msg_kernel<<<EE / 8, 256, 0, stream>>>(hcur, w, edge_index, agg, EE);
        update_kernel<<<(NN + 3) / 4, 256, 0, stream>>>(agg, deg, hcur, root, cbias, hnxt, NN);
        float* t = hcur; hcur = hnxt; hnxt = t;
    }

    // ---- output head ----
    head_kernel<<<(NN + 255) / 256, 256, 0, stream>>>(hcur, oW, ob, out, NN);
}
